// _NonLocalBlockND_76381698392744
// MI455X (gfx1250) — matrix operand order repaired and checked by reading, not compiled, not run on hardware
//
#include <hip/hip_runtime.h>

#define BB  8
#define CCH 256
#define ICN 128
#define OCN 256
#define HHd 64
#define NP  4096   /* 64*64 */
#define MPn 1024   /* pooled */

typedef __bf16 bf16_t;
typedef __attribute__((ext_vector_type(16))) __bf16 v16bf;
typedef __attribute__((ext_vector_type(8)))  float  v8f;

struct __align__(16) U4 { unsigned x, y, z, w; };
union FragU { v16bf v; U4 q[2]; };

// ---- CDNA5 async global->LDS path (hedged with __has_builtin) ----
#if defined(__has_builtin)
#  if __has_builtin(__builtin_amdgcn_global_load_async_to_lds_b128)
#    define HAVE_ASYNC_LDS 1
#  endif
#endif
#ifndef HAVE_ASYNC_LDS
#  define HAVE_ASYNC_LDS 0
#endif

typedef int v4i_vs __attribute__((vector_size(16)));
typedef __attribute__((address_space(1))) v4i_vs* gptr_v4i;
typedef __attribute__((address_space(3))) v4i_vs* lptr_v4i;

__device__ __forceinline__ void copy16_g2l(bf16_t* ldst, const bf16_t* gsrc) {
#if HAVE_ASYNC_LDS
  __builtin_amdgcn_global_load_async_to_lds_b128(
      (gptr_v4i)(size_t)gsrc,
      (lptr_v4i)(unsigned)(size_t)ldst, 0, 0);
#else
  *(U4*)ldst = *(const U4*)gsrc;
#endif
}
__device__ __forceinline__ void async_wait_all() {
#if HAVE_ASYNC_LDS
  asm volatile("s_wait_asynccnt 0x0" ::: "memory");
#endif
}

__device__ __forceinline__ bf16_t f2bf(float f) {
  union { float f; unsigned u; } a; a.f = f;
  unsigned u = a.u;
  unsigned r = u + 0x7FFFu + ((u >> 16) & 1u);
  union { unsigned short s; bf16_t b; } c; c.s = (unsigned short)(r >> 16);
  return c.b;
}
__device__ __forceinline__ float bf2f(bf16_t b) {
  union { unsigned short s; bf16_t b; } c; c.b = b;
  union { unsigned u; float f; } a; a.u = ((unsigned)c.s) << 16;
  return a.f;
}

// ---------------------------------------------------------------------------
// Generic bf16 WMMA GEMM: C[b] = A[b](MxK) * B[b](KxN), fp32 accumulate.
// Block tile 128x128, 8 waves, each wave 32(M) x 64(N) = 8 WMMAs / K-step.
// Double-buffered LDS; A staged via async-to-LDS; B staged transposed [n][k]
// with packed b32 stores.
// Epilogue: val = acc * cs * rowscale[b,m] * colscale[b,n] * bscale[b]
// M,N multiples of 128; K multiple of 32; ld* multiples of 8.
// ---------------------------------------------------------------------------
struct GemmArgs {
  const bf16_t* A;  long strideA; int lda;
  const bf16_t* Bm; long strideB; int ldb;
  float* Cf; bf16_t* Cb; long strideC; int ldc;
  int K;
  float cs;
  const float* rowscale; int rowstride;
  const float* colscale; int colstride;
  const float* bscale;
};

__global__ __launch_bounds__(256) void gemm_bf16_kernel(GemmArgs g) {
  __shared__ bf16_t   ldsA[2][128 * 32];   // [m][k] row-major
  __shared__ unsigned ldsB[2][128 * 16];   // [n][k-pairs] transposed, dword packed
  const int tid = threadIdx.x;
  const int b   = blockIdx.z;
  const int m0  = blockIdx.y * 128;
  const int n0  = blockIdx.x * 128;
  const bf16_t* A  = g.A  + (long)b * g.strideA;
  const bf16_t* Bm = g.Bm + (long)b * g.strideB;

  const int wid = tid >> 5, lane = tid & 31;
  const int wm  = wid & 3, ng = wid >> 2;    // wave tile: rows wm*32, cols ng*64
  const int r   = lane & 15, half = lane >> 4;

  // staging assignments
  const int arow = tid >> 1, acol = (tid & 1) * 16;     // A: 128 rows x 32 k
  const int k2   = (tid >> 4) * 2, nb = (tid & 15) * 8; // B: 32 k x 128 n

  v8f acc0[4] = {}; v8f acc1[4] = {};

  auto stage = [&](int buf, int kk) {
    const bf16_t* as = &A[(long)(m0 + arow) * g.lda + kk + acol];
    copy16_g2l(&ldsA[buf][arow * 32 + acol], as);
    copy16_g2l(&ldsA[buf][arow * 32 + acol + 8], as + 8);
    U4 r0 = *(const U4*)&Bm[(long)(kk + k2) * g.ldb + n0 + nb];
    U4 r1 = *(const U4*)&Bm[(long)(kk + k2 + 1) * g.ldb + n0 + nb];
    const unsigned short* p0 = (const unsigned short*)&r0;
    const unsigned short* p1 = (const unsigned short*)&r1;
    unsigned* bd = &ldsB[buf][0];
#pragma unroll
    for (int i = 0; i < 8; ++i)
      bd[(nb + i) * 16 + (k2 >> 1)] = (unsigned)p0[i] | ((unsigned)p1[i] << 16);
  };

  auto compute = [&](int buf) {
    FragU a0, a1, bf[4];
    const bf16_t* ab0 = &ldsA[buf][(wm * 32 + r) * 32];
    const bf16_t* ab1 = ab0 + 16 * 32;
    a0.q[0] = *(const U4*)(ab0 + half * 8);
    a0.q[1] = *(const U4*)(ab0 + 16 + half * 8);
    a1.q[0] = *(const U4*)(ab1 + half * 8);
    a1.q[1] = *(const U4*)(ab1 + 16 + half * 8);
#pragma unroll
    for (int nt = 0; nt < 4; ++nt) {
      const unsigned* bp = &ldsB[buf][(ng * 64 + nt * 16 + r) * 16 + half * 4];
      bf[nt].q[0] = ((const U4*)bp)[0];
      bf[nt].q[1] = ((const U4*)bp)[2];
    }
#pragma unroll
    for (int nt = 0; nt < 4; ++nt) {
      acc0[nt] = __builtin_amdgcn_wmma_f32_16x16x32_bf16(false, a0.v, false, bf[nt].v,
                                                         (short)0, acc0[nt], false, false);
      acc1[nt] = __builtin_amdgcn_wmma_f32_16x16x32_bf16(false, a1.v, false, bf[nt].v,
                                                         (short)0, acc1[nt], false, false);
    }
  };

  const int nsteps = g.K >> 5;
  stage(0, 0);
  async_wait_all();
  __syncthreads();
  for (int s = 0; s < nsteps; ++s) {
    const int cur = s & 1;
    if (s + 2 < nsteps) {
      __builtin_prefetch(&A[(long)(m0 + arow) * g.lda + (s + 2) * 32 + acol], 0, 1);
      __builtin_prefetch(&Bm[(long)((s + 2) * 32 + k2) * g.ldb + n0 + nb], 0, 1);
    }
    if (s + 1 < nsteps) stage(cur ^ 1, (s + 1) * 32);
    compute(cur);
    async_wait_all();
    __syncthreads();
  }

  float*  Cf = g.Cf ? g.Cf + (long)b * g.strideC : nullptr;
  bf16_t* Cb = g.Cb ? g.Cb + (long)b * g.strideC : nullptr;
#pragma unroll
  for (int mt = 0; mt < 2; ++mt) {
#pragma unroll
    for (int nt = 0; nt < 4; ++nt) {
#pragma unroll
      for (int v = 0; v < 8; ++v) {
        const int m = m0 + wm * 32 + mt * 16 + half * 8 + v;
        const int n = n0 + ng * 64 + nt * 16 + r;
        float val = (mt ? acc1[nt][v] : acc0[nt][v]) * g.cs;
        if (g.rowscale) val *= g.rowscale[(long)b * g.rowstride + m];
        if (g.colscale) val *= g.colscale[(long)b * g.colstride + n];
        if (g.bscale)   val *= g.bscale[b];
        const long idx = (long)m * g.ldc + n;
        if (Cb) Cb[idx] = f2bf(val); else Cf[idx] = val;
      }
    }
  }
}

// ---------------------------------------------------------------------------
// 3x3 partial conv as implicit GEMM with WMMA.
// Grid: (NP/128 pixel blocks (2 rows of 64), OC/128, B). Block tile 128x128.
// K = 9 taps * 256 channels, stepped 32 at a time; double-buffered.
// zin: (B, C, 66, 66) bf16 zero-padded; wtap: (9, OC, C) bf16.
// ---------------------------------------------------------------------------
__global__ __launch_bounds__(256) void conv3_wmma_kernel(const bf16_t* __restrict__ zin,
                                                         const bf16_t* __restrict__ wtap,
                                                         const float* __restrict__ inv3,
                                                         float* __restrict__ z) {
  __shared__ bf16_t   ldsA[2][128 * 32];
  __shared__ unsigned ldsB[2][128 * 16];
  const int tid = threadIdx.x;
  const int pb  = blockIdx.x;          // pixel block: rows h0 = pb*2, pb*2+1
  const int oc0 = blockIdx.y * 128;
  const int b   = blockIdx.z;
  const int h0  = pb * 2;
  const int wid = tid >> 5, lane = tid & 31;
  const int wm  = wid & 3, ng = wid >> 2;
  const int r   = lane & 15, half = lane >> 4;
  const int arow = tid >> 1, acol = (tid & 1) * 16;
  const int k2   = (tid >> 4) * 2, nb = (tid & 15) * 8;
  const int hh = nb >> 6, w0 = nb & 63;   // local pixel row / col for B staging

  v8f acc0[4] = {}; v8f acc1[4] = {};

  auto stage = [&](int buf, int s) {
    const int tap = s >> 3, kc = (s & 7) * 32;
    const int dy = tap / 3, dx = tap % 3;
    const bf16_t* as = &wtap[(long)tap * OCN * CCH + (long)(oc0 + arow) * CCH + kc + acol];
    copy16_g2l(&ldsA[buf][arow * 32 + acol], as);
    copy16_g2l(&ldsA[buf][arow * 32 + acol + 8], as + 8);
    const unsigned short* s0 = (const unsigned short*)
        (zin + ((long)(b * CCH + kc + k2) * 66 + (h0 + hh + dy)) * 66 + (w0 + dx));
    const unsigned short* s1 = s0 + 66 * 66;
    unsigned* bd = &ldsB[buf][0];
#pragma unroll
    for (int i = 0; i < 8; ++i)
      bd[(nb + i) * 16 + (k2 >> 1)] = (unsigned)s0[i] | ((unsigned)s1[i] << 16);
  };

  auto compute = [&](int buf) {
    FragU a0, a1, bf[4];
    const bf16_t* ab0 = &ldsA[buf][(wm * 32 + r) * 32];
    const bf16_t* ab1 = ab0 + 16 * 32;
    a0.q[0] = *(const U4*)(ab0 + half * 8);
    a0.q[1] = *(const U4*)(ab0 + 16 + half * 8);
    a1.q[0] = *(const U4*)(ab1 + half * 8);
    a1.q[1] = *(const U4*)(ab1 + 16 + half * 8);
#pragma unroll
    for (int nt = 0; nt < 4; ++nt) {
      const unsigned* bp = &ldsB[buf][(ng * 64 + nt * 16 + r) * 16 + half * 4];
      bf[nt].q[0] = ((const U4*)bp)[0];
      bf[nt].q[1] = ((const U4*)bp)[2];
    }
#pragma unroll
    for (int nt = 0; nt < 4; ++nt) {
      acc0[nt] = __builtin_amdgcn_wmma_f32_16x16x32_bf16(false, a0.v, false, bf[nt].v,
                                                         (short)0, acc0[nt], false, false);
      acc1[nt] = __builtin_amdgcn_wmma_f32_16x16x32_bf16(false, a1.v, false, bf[nt].v,
                                                         (short)0, acc1[nt], false, false);
    }
  };

  stage(0, 0);
  async_wait_all();
  __syncthreads();
  for (int s = 0; s < 72; ++s) {         // 9 taps * 8 K-chunks
    const int cur = s & 1;
    if (s + 1 < 72) stage(cur ^ 1, s + 1);
    compute(cur);
    async_wait_all();
    __syncthreads();
  }

  const float* ip = inv3 + (long)b * NP + pb * 128;
#pragma unroll
  for (int mt = 0; mt < 2; ++mt) {
#pragma unroll
    for (int nt = 0; nt < 4; ++nt) {
#pragma unroll
      for (int v = 0; v < 8; ++v) {
        const int m = oc0 + wm * 32 + mt * 16 + half * 8 + v;
        const int n = ng * 64 + nt * 16 + r;
        const float val = (mt ? acc1[nt][v] : acc0[nt][v]) * ip[n];
        z[((long)b * OCN + m) * NP + pb * 128 + n] = val;
      }
    }
  }
}

// ---------------------------------------------------------------------------
// Elementwise / reduction helpers
// ---------------------------------------------------------------------------
__device__ __forceinline__ void blk_reduce2(float& s, float& s2) {
  __shared__ float sh[256], sh2[256];
  int t = threadIdx.x;
  sh[t] = s; sh2[t] = s2; __syncthreads();
  for (int o = 128; o > 0; o >>= 1) {
    if (t < o) { sh[t] += sh[t + o]; sh2[t] += sh2[t + o]; }
    __syncthreads();
  }
  s = sh[0]; s2 = sh2[0];
}

__global__ void k_prep(const float* __restrict__ x, const float* __restrict__ mask,
                       bf16_t* __restrict__ xm, float* __restrict__ inv1,
                       float* __restrict__ tmask) {
  const int p = blockIdx.x * 256 + threadIdx.x;
  const int b = blockIdx.y;
  float s = 0.f;
  for (int c = 0; c < CCH; ++c) {
    const long idx = ((long)(b * CCH + c)) * NP + p;
    const float mv = mask[idx];
    xm[idx] = f2bf(x[idx] * mv);
    s += mv;
  }
  const long pi = (long)b * NP + p;
  inv1[pi]  = (s == 0.f) ? 0.f : (1.f / s);
  tmask[pi] = (s == 0.f) ? 0.f : 1.f;
}

__global__ void k_wcat(const float* th, const float* ph, const float* gw, bf16_t* wcat) {
  const int i = blockIdx.x * 256 + threadIdx.x;
  if (i >= 384 * CCH) return;
  const int rrow = i >> 8, cc = i & 255;
  const float* src = (rrow < 128) ? th : ((rrow < 256) ? ph : gw);
  wcat[i] = f2bf(src[(rrow & 127) * CCH + cc]);
}

__global__ void k_wt(const float* Ww, bf16_t* wt) {
  const int i = blockIdx.x * 256 + threadIdx.x;
  if (i >= ICN * CCH) return;
  const int ic = i >> 8, cc = i & 255;
  wt[i] = f2bf(Ww[cc * ICN + ic]);
}

__global__ void k_wtap(const float* pw, bf16_t* wtap) {
  const int i = blockIdx.x * 256 + threadIdx.x;
  if (i >= 9 * OCN * CCH) return;
  const int tap = i / (OCN * CCH);
  const int rem = i % (OCN * CCH);
  const int oc = rem >> 8, cc = rem & 255;
  wtap[i] = f2bf(pw[((long)(oc * CCH + cc)) * 9 + tap]);
}

__global__ void k_theta_t(const bf16_t* __restrict__ raw1, bf16_t* __restrict__ theta_t) {
  const long i = (long)blockIdx.x * 256 + threadIdx.x;
  const int ic = (int)(i & 127);
  const int n  = (int)((i >> 7) & 4095);
  const int b  = (int)(i >> 19);
  theta_t[i] = raw1[((long)b * 384 + ic) * NP + n];
}

__global__ void k_pool(const bf16_t* __restrict__ raw1, const float* __restrict__ tmask,
                       bf16_t* __restrict__ phi, bf16_t* __restrict__ g_t,
                       float* __restrict__ mp) {
  const long i = (long)blockIdx.x * 256 + threadIdx.x;
  const int m  = (int)(i & 1023);
  const int ic = (int)((i >> 10) & 127);
  const int b  = (int)(i >> 17);
  const int my = m >> 5, mx = m & 31;
  const int p00 = (2 * my) * 64 + 2 * mx;
  const long bphi = ((long)b * 384 + 128 + ic) * NP;
  const long bg   = ((long)b * 384 + 256 + ic) * NP;
  float pv = fmaxf(fmaxf(bf2f(raw1[bphi + p00]),      bf2f(raw1[bphi + p00 + 1])),
                   fmaxf(bf2f(raw1[bphi + p00 + 64]), bf2f(raw1[bphi + p00 + 65])));
  float gv = fmaxf(fmaxf(bf2f(raw1[bg + p00]),      bf2f(raw1[bg + p00 + 1])),
                   fmaxf(bf2f(raw1[bg + p00 + 64]), bf2f(raw1[bg + p00 + 65])));
  const long tb = (long)b * NP;
  const float mpv = fmaxf(fmaxf(tmask[tb + p00], tmask[tb + p00 + 1]),
                          fmaxf(tmask[tb + p00 + 64], tmask[tb + p00 + 65]));
  phi[((long)b * ICN + ic) * MPn + m] = f2bf(pv * mpv);
  g_t[((long)b * MPn + m) * ICN + ic] = f2bf(gv);
  if (ic == 0) mp[(long)b * MPn + m] = mpv;
}

__global__ void k_sb2(const float* __restrict__ mp, float* __restrict__ sb2) {
  const int b = blockIdx.x;
  float s = 0.f, s2 = 0.f;
  for (int j = threadIdx.x; j < MPn; j += 256) s += mp[(long)b * MPn + j];
  blk_reduce2(s, s2);
  if (threadIdx.x == 0) sb2[b] = 128.f * s;
}

__global__ void k_inv2(const float* __restrict__ tmask, const float* __restrict__ sb2,
                       float* __restrict__ inv2, float* __restrict__ wym) {
  const long i = (long)blockIdx.x * 256 + threadIdx.x;
  const int b = (int)(i >> 12);
  const float ms2 = 128.f * tmask[i] * sb2[b];
  inv2[i] = (ms2 == 0.f) ? 0.f : (1.f / ms2);
  wym[i]  = (ms2 == 0.f) ? 0.f : 1.f;
}

__global__ void k_in_stats_strided(const float* __restrict__ wy, float* mean, float* rstd) {
  const int c = blockIdx.x, b = blockIdx.y;
  float s = 0.f, s2 = 0.f;
  for (int n = threadIdx.x; n < NP; n += 256) {
    const float v = wy[((long)b * NP + n) * CCH + c];
    s += v; s2 += v * v;
  }
  blk_reduce2(s, s2);
  if (threadIdx.x == 0) {
    const float m = s / NP;
    mean[b * CCH + c] = m;
    rstd[b * CCH + c] = rsqrtf(s2 / NP - m * m + 1e-5f);
  }
}

__global__ void k_in_stats_contig(const float* __restrict__ z, float* mean, float* rstd) {
  const int c = blockIdx.x, b = blockIdx.y;
  float s = 0.f, s2 = 0.f;
  for (int n = threadIdx.x; n < NP; n += 256) {
    const float v = z[((long)b * CCH + c) * NP + n];
    s += v; s2 += v * v;
  }
  blk_reduce2(s, s2);
  if (threadIdx.x == 0) {
    const float m = s / NP;
    mean[b * CCH + c] = m;
    rstd[b * CCH + c] = rsqrtf(s2 / NP - m * m + 1e-5f);
  }
}

__global__ void k_zms(const float* __restrict__ mask, const float* __restrict__ wym,
                      float* __restrict__ zms) {
  const int p = blockIdx.x * 256 + threadIdx.x;
  const int b = blockIdx.y;
  const long pi = (long)b * NP + p;
  float cnt;
  if (wym[pi] != 0.f) {
    cnt = (float)CCH;
  } else {
    int k = 0;
    for (int c = 0; c < CCH; ++c)
      k += (mask[((long)(b * CCH + c)) * NP + p] != 0.f) ? 1 : 0;
    cnt = (float)k;
  }
  zms[pi] = cnt;
}

__global__ void k_zero_u32(unsigned* __restrict__ p, long n) {
  const long i = (long)blockIdx.x * 256 + threadIdx.x;
  if (i < n) p[i] = 0u;
}

__global__ void k_zin(const float* __restrict__ wy, const float* __restrict__ mean1,
                      const float* __restrict__ rstd1, const float* __restrict__ x,
                      const float* __restrict__ mask, const float* __restrict__ wym,
                      bf16_t* __restrict__ zin) {
  const long i = (long)blockIdx.x * 256 + threadIdx.x;
  const int p = (int)(i & 4095);
  const int c = (int)((i >> 12) & 255);
  const int b = (int)(i >> 20);
  const float wyv = wy[((long)b * NP + p) * CCH + c];
  const float zn  = (wyv - mean1[b * CCH + c]) * rstd1[b * CCH + c];
  const float zi  = zn + x[i];
  const long pi = (long)b * NP + p;
  const float zb = ((mask[i] != 0.f) || (wym[pi] != 0.f)) ? 1.f : 0.f;
  const int h = p >> 6, w = p & 63;
  zin[((long)(b * CCH + c) * 66 + (h + 1)) * 66 + (w + 1)] = f2bf(zi * zb);
}

__global__ void k_ms3(const float* __restrict__ zms, float* __restrict__ inv3,
                      float* __restrict__ nmv) {
  const long i = (long)blockIdx.x * 256 + threadIdx.x;
  const int p = (int)(i & 4095);
  const int b = (int)(i >> 12);
  const int h = p >> 6, w = p & 63;
  float s = 0.f;
  for (int dy = -1; dy <= 1; ++dy)
    for (int dx = -1; dx <= 1; ++dx) {
      const int hh = h + dy, ww = w + dx;
      if (hh >= 0 && hh < 64 && ww >= 0 && ww < 64)
        s += zms[(long)b * NP + hh * 64 + ww];
    }
  inv3[i] = (s == 0.f) ? 0.f : (1.f / s);
  nmv[i]  = (s == 0.f) ? 0.f : 1.f;
}

__global__ void k_final(const float* __restrict__ z, const float* __restrict__ mean2,
                        const float* __restrict__ rstd2, const float* __restrict__ nmv,
                        float* __restrict__ out) {
  const long i = (long)blockIdx.x * 256 + threadIdx.x;
  const int p  = (int)(i & 4095);
  const int oc = (int)((i >> 12) & 255);
  const int b  = (int)(i >> 20);
  float o = (z[i] - mean2[b * OCN + oc]) * rstd2[b * OCN + oc];
  o = (o < 0.f) ? 0.2f * o : o;
  out[i] = o;
  out[(long)BB * OCN * NP + i] = nmv[(long)b * NP + p];
}

// ---------------------------------------------------------------------------
extern "C" void kernel_launch(void* const* d_in, const int* in_sizes, int n_in,
                              void* d_out, int out_size, void* d_ws, size_t ws_size,
                              hipStream_t stream) {
  (void)in_sizes; (void)n_in; (void)out_size;
  const float* x    = (const float*)d_in[0];
  const float* mask = (const float*)d_in[1];
  const float* thw  = (const float*)d_in[2];
  const float* phw  = (const float*)d_in[4];
  const float* gw   = (const float*)d_in[6];
  const float* Ww   = (const float*)d_in[8];
  const float* pw   = (const float*)d_in[10];
  float* out = (float*)d_out;

  char* w = (char*)d_ws;
  size_t off = 0;
  auto alloc = [&](size_t bytes) -> char* {
    char* p = w + off;
    off = (off + bytes + 255) & ~(size_t)255;
    return p;
  };

  float*  inv1    = (float*) alloc((size_t)BB * NP * 4);
  float*  tmask   = (float*) alloc((size_t)BB * NP * 4);
  bf16_t* xm      = (bf16_t*)alloc((size_t)BB * CCH * NP * 2);
  bf16_t* wcat    = (bf16_t*)alloc((size_t)384 * CCH * 2);
  bf16_t* wt      = (bf16_t*)alloc((size_t)ICN * CCH * 2);
  bf16_t* wtap    = (bf16_t*)alloc((size_t)9 * OCN * CCH * 2);
  bf16_t* raw1    = (bf16_t*)alloc((size_t)BB * 384 * NP * 2);
  bf16_t* theta_t = (bf16_t*)alloc((size_t)BB * NP * ICN * 2);
  bf16_t* phi     = (bf16_t*)alloc((size_t)BB * ICN * MPn * 2);
  bf16_t* g_t     = (bf16_t*)alloc((size_t)BB * MPn * ICN * 2);
  float*  mp      = (float*) alloc((size_t)BB * MPn * 4);
  float*  sb2     = (float*) alloc((size_t)BB * 4);
  bf16_t* fdc     = (bf16_t*)alloc((size_t)BB * NP * MPn * 2);
  bf16_t* y_t     = (bf16_t*)alloc((size_t)BB * NP * ICN * 2);
  float*  inv2    = (float*) alloc((size_t)BB * NP * 4);
  float*  wym     = (float*) alloc((size_t)BB * NP * 4);
  float*  wy      = (float*) alloc((size_t)BB * NP * CCH * 4);
  float*  mean1   = (float*) alloc((size_t)BB * CCH * 4);
  float*  rstd1   = (float*) alloc((size_t)BB * CCH * 4);
  bf16_t* zin     = (bf16_t*)alloc((size_t)BB * CCH * 66 * 66 * 2);
  float*  zms     = (float*) alloc((size_t)BB * NP * 4);
  float*  inv3    = (float*) alloc((size_t)BB * NP * 4);
  float*  nmv     = (float*) alloc((size_t)BB * NP * 4);
  float*  z       = (float*) alloc((size_t)BB * OCN * NP * 4);
  float*  mean2   = (float*) alloc((size_t)BB * OCN * 4);
  float*  rstd2   = (float*) alloc((size_t)BB * OCN * 4);
  if (off > ws_size) return;

  k_prep<<<dim3(NP / 256, BB), 256, 0, stream>>>(x, mask, xm, inv1, tmask);
  k_wcat<<<(384 * CCH + 255) / 256, 256, 0, stream>>>(thw, phw, gw, wcat);
  k_wt<<<(ICN * CCH + 255) / 256, 256, 0, stream>>>(Ww, wt);
  k_wtap<<<(9 * OCN * CCH + 255) / 256, 256, 0, stream>>>(pw, wtap);

  // fused theta/phi/g 1x1 partial conv: raw1[b] = wcat(384x256) * xm[b](256x4096)
  {
    GemmArgs g{};
    g.A = wcat;  g.strideA = 0;              g.lda = CCH;
    g.Bm = xm;   g.strideB = (long)CCH * NP; g.ldb = NP;
    g.Cf = nullptr; g.Cb = raw1; g.strideC = (long)384 * NP; g.ldc = NP;
    g.K = CCH; g.cs = 1.f;
    g.rowscale = nullptr; g.rowstride = 0;
    g.colscale = inv1;    g.colstride = NP;
    g.bscale = nullptr;
    gemm_bf16_kernel<<<dim3(NP / 128, 384 / 128, BB), 256, 0, stream>>>(g);
  }

  k_theta_t<<<(int)(((long)BB * NP * ICN) / 256), 256, 0, stream>>>(raw1, theta_t);
  k_pool<<<(int)(((long)BB * ICN * MPn) / 256), 256, 0, stream>>>(raw1, tmask, phi, g_t, mp);
  k_sb2<<<BB, 256, 0, stream>>>(mp, sb2);

  // f GEMM: fdc[b](4096x1024) = theta_t[b](4096x128) * phi[b](128x1024) * 0.125
  {
    GemmArgs g{};
    g.A = theta_t; g.strideA = (long)NP * ICN;  g.lda = ICN;
    g.Bm = phi;    g.strideB = (long)ICN * MPn; g.ldb = MPn;
    g.Cf = nullptr; g.Cb = fdc; g.strideC = (long)NP * MPn; g.ldc = MPn;
    g.K = ICN; g.cs = 0.125f;
    g.rowscale = nullptr; g.colscale = nullptr; g.bscale = nullptr;
    gemm_bf16_kernel<<<dim3(MPn / 128, NP / 128, BB), 256, 0, stream>>>(g);
  }

  // y GEMM: y_t[b](4096x128) = fdc[b](4096x1024) * g_t[b](1024x128) * t[n] * sb2[b]
  {
    GemmArgs g{};
    g.A = fdc;  g.strideA = (long)NP * MPn;  g.lda = MPn;
    g.Bm = g_t; g.strideB = (long)MPn * ICN; g.ldb = ICN;
    g.Cf = nullptr; g.Cb = y_t; g.strideC = (long)NP * ICN; g.ldc = ICN;
    g.K = MPn; g.cs = 1.f;
    g.rowscale = tmask; g.rowstride = NP;
    g.colscale = nullptr;
    g.bscale = sb2;
    gemm_bf16_kernel<<<dim3(ICN / 128, NP / 128, BB), 256, 0, stream>>>(g);
  }

  // W 1x1 partial conv: wy[b](4096x256) = y_t[b](4096x128) * wt(128x256) / ms2
  k_inv2<<<(int)(((long)BB * NP) / 256), 256, 0, stream>>>(tmask, sb2, inv2, wym);
  {
    GemmArgs g{};
    g.A = y_t;  g.strideA = (long)NP * ICN; g.lda = ICN;
    g.Bm = wt;  g.strideB = 0;              g.ldb = CCH;
    g.Cf = wy; g.Cb = nullptr; g.strideC = (long)NP * CCH; g.ldc = CCH;
    g.K = ICN; g.cs = 1.f;
    g.rowscale = inv2; g.rowstride = NP;
    g.colscale = nullptr; g.bscale = nullptr;
    gemm_bf16_kernel<<<dim3(CCH / 128, NP / 128, BB), 256, 0, stream>>>(g);
  }

  k_in_stats_strided<<<dim3(CCH, BB), 256, 0, stream>>>(wy, mean1, rstd1);
  k_zms<<<dim3(NP / 256, BB), 256, 0, stream>>>(mask, wym, zms);
  {
    const long nzero = (long)BB * CCH * 66 * 66 / 2;
    k_zero_u32<<<(int)((nzero + 255) / 256), 256, 0, stream>>>((unsigned*)zin, nzero);
  }
  k_zin<<<(int)(((long)BB * CCH * NP) / 256), 256, 0, stream>>>(wy, mean1, rstd1, x,
                                                               mask, wym, zin);
  k_ms3<<<(int)(((long)BB * NP) / 256), 256, 0, stream>>>(zms, inv3, nmv);

  conv3_wmma_kernel<<<dim3(NP / 128, OCN / 128, BB), 256, 0, stream>>>(zin, wtap, inv3, z);

  k_in_stats_contig<<<dim3(OCN, BB), 256, 0, stream>>>(z, mean2, rstd2);
  k_final<<<(int)(((long)BB * OCN * NP) / 256), 256, 0, stream>>>(z, mean2, rstd2, nmv, out);
}
